// Graph_encoder_23003844837986
// MI455X (gfx1250) — matrix operand order repaired and checked by reading, not compiled, not run on hardware
//
#include <hip/hip_runtime.h>
#include <hip/hip_bf16.h>
#include <math.h>

#define NN 65536   // nodes (B*L)
#define NE 262144  // edges
#define HD 256     // hidden

typedef __bf16 bf16_t;
typedef bf16_t v16bf __attribute__((ext_vector_type(16)));
typedef bf16_t v4bf  __attribute__((ext_vector_type(4)));
typedef bf16_t v2bf  __attribute__((ext_vector_type(2)));
typedef float  v8f   __attribute__((ext_vector_type(8)));
typedef unsigned int u32x4 __attribute__((ext_vector_type(4)));
typedef int i32x4 __attribute__((ext_vector_type(4)));
typedef int i32x8 __attribute__((ext_vector_type(8)));

static constexpr int LDA = 34;  // LDS row stride in bf16 elems (64B row + 4B TDM pad)

// ---------------- WMMA helpers (CDNA5 16x16x32 bf16, f32 accum) ----------------

__device__ __forceinline__ v8f wmma_bf16(v16bf a, v16bf b, v8f c) {
  // 8 args: (neg_a, A, neg_b, B, c_mod, C, reuse_a, reuse_b)
  return __builtin_amdgcn_wmma_f32_16x16x32_bf16(false, a, false, b, (short)0, c,
                                                 false, false);
}

// A fragment (16x32 bf16, MxK) from LDS tile (row-major, stride LDA).
__device__ __forceinline__ v16bf load_a_frag(const bf16_t* sA, int lane) {
  const int row = lane & 15, half = lane >> 4;
  v16bf a{};
#pragma unroll
  for (int v = 0; v < 8; ++v) {
    const int k0 = 2 * v + (v >= 4 ? 8 : 0) + half * 8;
    v2bf p = *(const v2bf*)(sA + row * LDA + k0);
    a[2 * v] = p.x;
    a[2 * v + 1] = p.y;
  }
  return a;
}

// B fragment (32x16 bf16, KxN) for out = A @ W^T, W row-major [nrows][ldk].
__device__ __forceinline__ v16bf load_b_frag(const bf16_t* __restrict__ W, int ldk,
                                             int n0, int kbase, int lane) {
  const int n = n0 + (lane & 15);
  const bf16_t* p = W + (size_t)n * ldk + kbase + ((lane >> 4) << 3);
  v16bf b{};
#pragma unroll
  for (int v = 0; v < 8; ++v) {
    v2bf q = *(const v2bf*)(p + 2 * v + (v >= 4 ? 8 : 0));
    b[2 * v] = q.x;
    b[2 * v + 1] = q.y;
  }
  return b;
}

// ---------------- Tensor Data Mover: 64 rows x 32 bf16 tile -> LDS ----------------
// D# per CDNA5 ISA ch.8: group0 = {flags, lds_addr, global_addr}, group1 = dims.
// pad_enable inserts 1 DWORD after every 16 DWORDs (64B row) -> LDS stride 34 elems.
__device__ __forceinline__ void tdm_load_tile(const bf16_t* gsrc, unsigned int lds_off) {
  const unsigned long long ga = (unsigned long long)(size_t)gsrc;
  u32x4 g0;
  g0[0] = 1u;                                                    // count=1 (user D#)
  g0[1] = lds_off;                                               // lds_addr (bytes)
  g0[2] = (unsigned int)(ga & 0xFFFFFFFFu);                      // global_addr[31:0]
  g0[3] = (unsigned int)((ga >> 32) & 0x01FFFFFFu) | (2u << 30); // addr[56:32], type=2
  const unsigned int td0 = HD;   // tensor_dim0 (elems per row)
  const unsigned int td1 = NN;   // tensor_dim1 (rows)
  i32x8 g1;
  g1[0] = (int)((1u << 16) |     // data_size = 2 bytes
                (1u << 20) |     // pad_enable
                (3u << 22));     // pad_interval: 16 DWORDs (pad_amount=0 -> 1 DWORD)
  g1[1] = (int)((td0 & 0xFFFFu) << 16);                     // abar=0 | tensor_dim0.lo
  g1[2] = (int)((td0 >> 16) | ((td1 & 0xFFFFu) << 16));     // tensor_dim0.hi | dim1.lo
  g1[3] = (int)((td1 >> 16) | (32u << 16));                 // tensor_dim1.hi | tile_dim0=32
  g1[4] = (int)64u;                                         // tile_dim1=64, tile_dim2=0
  g1[5] = (int)HD;                                          // tensor_dim0_stride.lo
  g1[6] = 0;                                                // stride0.hi | stride1.lo
  g1[7] = 0;
  const i32x4 z4 = {0, 0, 0, 0};
#if __clang_major__ >= 23
  const i32x8 z8 = {0, 0, 0, 0, 0, 0, 0, 0};
  __builtin_amdgcn_tensor_load_to_lds(g0, g1, z4, z4, z8, 0);
#else
  __builtin_amdgcn_tensor_load_to_lds(g0, g1, z4, z4, 0);
#endif
}

__device__ __forceinline__ float sigm(float x) { return 1.0f / (1.0f + __expf(-x)); }

// ---------------- small prep kernels ----------------

__global__ __launch_bounds__(256) void cvt_f32_to_bf16(const float* __restrict__ in,
                                                       bf16_t* __restrict__ out, int n4) {
  int i = blockIdx.x * 256 + threadIdx.x;
  if (i < n4) {
    float4 v = ((const float4*)in)[i];
    v4bf o;
    o.x = (bf16_t)v.x; o.y = (bf16_t)v.y; o.z = (bf16_t)v.z; o.w = (bf16_t)v.w;
    ((v4bf*)out)[i] = o;
  }
}

__global__ __launch_bounds__(256) void bias_sum(const float* __restrict__ a,
                                                const float* __restrict__ b,
                                                float* __restrict__ o) {
  int i = blockIdx.x * 256 + threadIdx.x;
  if (i < 4 * HD) o[i] = a[i] + b[i];
}

__global__ __launch_bounds__(256) void deg_init(float* __restrict__ deg) {
  int i = blockIdx.x * 256 + threadIdx.x;
  if (i < NN) deg[i] = 1.0f;  // self-loop weight
}

__global__ __launch_bounds__(256) void deg_edges(const int* __restrict__ dst,
                                                 const float* __restrict__ ew,
                                                 float* __restrict__ deg) {
  int e = blockIdx.x * 256 + threadIdx.x;
  if (e < NE) unsafeAtomicAdd(&deg[dst[e]], ew[e]);
}

__global__ __launch_bounds__(256) void deg_fin(const float* __restrict__ deg,
                                               float* __restrict__ dinv) {
  int i = blockIdx.x * 256 + threadIdx.x;
  if (i < NN) {
    float d = deg[i];
    dinv[i] = d > 0.0f ? rsqrtf(d) : 0.0f;
  }
}

// out[i,:] = bias + dinv[i]^2 * h[i,:]   (self loop fused with bias)
__global__ __launch_bounds__(256) void scatter_init(const float* __restrict__ h,
                                                    const float* __restrict__ dinv,
                                                    const float* __restrict__ bias,
                                                    float* __restrict__ out) {
  size_t i = (size_t)blockIdx.x * 256 + threadIdx.x;  // over NN*HD
  int node = (int)(i >> 8), c = (int)(i & 255);
  float dv = dinv[node];
  out[i] = bias[c] + dv * dv * h[i];
}

// edge-parallel scatter: 64 lanes/edge, float4 per lane, hw f32 atomics into L2
__global__ __launch_bounds__(256) void scatter_edges(const int* __restrict__ src,
                                                     const int* __restrict__ dst,
                                                     const float* __restrict__ ew,
                                                     const float* __restrict__ dinv,
                                                     const float* __restrict__ h,
                                                     float* __restrict__ out) {
  int t = blockIdx.x * 256 + threadIdx.x;
  int e = t >> 6;
  int j = (t & 63) * 4;
  int s = src[e], d = dst[e];
  float norm = dinv[s] * ew[e] * dinv[d];
  float4 hv = *(const float4*)(h + (size_t)s * HD + j);
  float* op = out + (size_t)d * HD + j;
  unsafeAtomicAdd(op + 0, norm * hv.x);
  unsafeAtomicAdd(op + 1, norm * hv.y);
  unsafeAtomicAdd(op + 2, norm * hv.z);
  unsafeAtomicAdd(op + 3, norm * hv.w);
}

// ---------------- WMMA GEMM: out[N][HD] = A(bf16) @ W(bf16 [HD][HD])^T
// 64x32 tile / block, 8 wave32; A tiles staged by TDM, double-buffered LDS.
__global__ __launch_bounds__(256) void gemm_lin(const bf16_t* __restrict__ A,
                                                const bf16_t* __restrict__ W,
                                                float* __restrict__ out) {
  __shared__ bf16_t sA[2][64 * LDA];
  const int tid = threadIdx.x, lane = tid & 31, wv = tid >> 5;
  const int rt = wv & 3, ct = wv >> 2;
  const int m0 = blockIdx.x * 64, n0 = blockIdx.y * 32;
  const unsigned int lds0 = (unsigned int)(unsigned long long)(size_t)&sA[0][0];
  const unsigned int lds1 = (unsigned int)(unsigned long long)(size_t)&sA[1][0];
  constexpr int NK = HD / 32;
  if (wv == 0) tdm_load_tile(A + (size_t)m0 * HD, lds0);
  v8f acc = {};
  for (int kc = 0; kc < NK; ++kc) {
    if (wv == 0) {
      if (kc + 1 < NK) {
        tdm_load_tile(A + (size_t)m0 * HD + (kc + 1) * 32, ((kc + 1) & 1) ? lds1 : lds0);
        __builtin_amdgcn_s_wait_tensorcnt((short)1);  // current buffer resident
      } else {
        __builtin_amdgcn_s_wait_tensorcnt((short)0);
      }
    }
    __syncthreads();  // release compute waves on current buffer
    const bf16_t* cur = sA[kc & 1];
    v16bf a = load_a_frag(cur + rt * 16 * LDA, lane);
    v16bf b = load_b_frag(W, HD, n0 + ct * 16, kc * 32, lane);
    acc = wmma_bf16(a, b, acc);
    __syncthreads();  // all done reading before buffer is re-filled
  }
  const int n = n0 + ct * 16 + (lane & 15);
  const int mb = m0 + rt * 16 + ((lane >> 4) << 3);
#pragma unroll
  for (int r = 0; r < 8; ++r) out[(size_t)(mb + r) * HD + n] = acc[r];
}

// ---------------- LSTM step 1: h0=c0=0 -> only i,g,o gates needed.
__global__ __launch_bounds__(256) void lstm_step1(const bf16_t* __restrict__ X,
                                                  const bf16_t* __restrict__ Wih,
                                                  const float* __restrict__ bb,
                                                  float* __restrict__ c1,
                                                  bf16_t* __restrict__ h1) {
  __shared__ bf16_t sA[2][64 * LDA];
  const int tid = threadIdx.x, lane = tid & 31, wv = tid >> 5;
  const int rt = wv & 3, ct = wv >> 2;
  const int m0 = blockIdx.x * 64, u0 = blockIdx.y * 32;
  const int nc = u0 + ct * 16;
  const unsigned int lds0 = (unsigned int)(unsigned long long)(size_t)&sA[0][0];
  const unsigned int lds1 = (unsigned int)(unsigned long long)(size_t)&sA[1][0];
  constexpr int NK = HD / 32;
  if (wv == 0) tdm_load_tile(X + (size_t)m0 * HD, lds0);
  v8f ai = {}, ag = {}, ao = {};
  for (int kc = 0; kc < NK; ++kc) {
    if (wv == 0) {
      if (kc + 1 < NK) {
        tdm_load_tile(X + (size_t)m0 * HD + (kc + 1) * 32, ((kc + 1) & 1) ? lds1 : lds0);
        __builtin_amdgcn_s_wait_tensorcnt((short)1);
      } else {
        __builtin_amdgcn_s_wait_tensorcnt((short)0);
      }
    }
    __syncthreads();
    const bf16_t* cur = sA[kc & 1];
    v16bf a = load_a_frag(cur + rt * 16 * LDA, lane);
    v16bf bi = load_b_frag(Wih, HD, 0 * HD + nc, kc * 32, lane);
    v16bf bg = load_b_frag(Wih, HD, 2 * HD + nc, kc * 32, lane);
    v16bf bo = load_b_frag(Wih, HD, 3 * HD + nc, kc * 32, lane);
    ai = wmma_bf16(a, bi, ai);
    ag = wmma_bf16(a, bg, ag);
    ao = wmma_bf16(a, bo, ao);
    __syncthreads();
  }
  const int n = nc + (lane & 15);
  const int mb = m0 + rt * 16 + ((lane >> 4) << 3);
  const float b_i = bb[n], b_g = bb[2 * HD + n], b_o = bb[3 * HD + n];
#pragma unroll
  for (int r = 0; r < 8; ++r) {
    float cc = sigm(ai[r] + b_i) * tanhf(ag[r] + b_g);
    float hh = sigm(ao[r] + b_o) * tanhf(cc);
    size_t idx = (size_t)(mb + r) * HD + n;
    c1[idx] = cc;
    h1[idx] = (bf16_t)hh;
  }
}

// ---------------- LSTM step 2: z = Xm@Wih^T + H1@Whh^T + bb (K=512 split loop)
__global__ __launch_bounds__(256) void lstm_step2(const bf16_t* __restrict__ Xm,
                                                  const bf16_t* __restrict__ H1,
                                                  const bf16_t* __restrict__ Wih,
                                                  const bf16_t* __restrict__ Whh,
                                                  const float* __restrict__ bb,
                                                  const float* __restrict__ c1,
                                                  bf16_t* __restrict__ out_bf,
                                                  float* __restrict__ out_f,
                                                  const float* __restrict__ mask) {
  __shared__ bf16_t sA[2][64 * LDA];
  const int tid = threadIdx.x, lane = tid & 31, wv = tid >> 5;
  const int rt = wv & 3, ct = wv >> 2;
  const int m0 = blockIdx.x * 64, u0 = blockIdx.y * 32;
  const int nc = u0 + ct * 16;
  const unsigned int lds0 = (unsigned int)(unsigned long long)(size_t)&sA[0][0];
  const unsigned int lds1 = (unsigned int)(unsigned long long)(size_t)&sA[1][0];
  constexpr int NK = 16;
  if (wv == 0) tdm_load_tile(Xm + (size_t)m0 * HD, lds0);
  v8f ai = {}, af = {}, ag = {}, ao = {};
  for (int kc = 0; kc < NK; ++kc) {
    if (wv == 0) {
      if (kc + 1 < NK) {
        const bf16_t* nsrc = (kc + 1 < 8) ? Xm : H1;
        tdm_load_tile(nsrc + (size_t)m0 * HD + ((kc + 1) & 7) * 32,
                      ((kc + 1) & 1) ? lds1 : lds0);
        __builtin_amdgcn_s_wait_tensorcnt((short)1);
      } else {
        __builtin_amdgcn_s_wait_tensorcnt((short)0);
      }
    }
    __syncthreads();
    const bf16_t* cur = sA[kc & 1];
    const bf16_t* Wsrc = (kc < 8) ? Wih : Whh;
    const int ko = (kc & 7) * 32;
    v16bf a = load_a_frag(cur + rt * 16 * LDA, lane);
    v16bf bi = load_b_frag(Wsrc, HD, 0 * HD + nc, ko, lane);
    v16bf bf_ = load_b_frag(Wsrc, HD, 1 * HD + nc, ko, lane);
    v16bf bg = load_b_frag(Wsrc, HD, 2 * HD + nc, ko, lane);
    v16bf bo = load_b_frag(Wsrc, HD, 3 * HD + nc, ko, lane);
    ai = wmma_bf16(a, bi, ai);
    af = wmma_bf16(a, bf_, af);
    ag = wmma_bf16(a, bg, ag);
    ao = wmma_bf16(a, bo, ao);
    __syncthreads();
  }
  const int n = nc + (lane & 15);
  const int mb = m0 + rt * 16 + ((lane >> 4) << 3);
  const float b_i = bb[n], b_f = bb[HD + n], b_g = bb[2 * HD + n], b_o = bb[3 * HD + n];
#pragma unroll
  for (int r = 0; r < 8; ++r) {
    size_t idx = (size_t)(mb + r) * HD + n;
    float cp = c1[idx];
    float cc = sigm(af[r] + b_f) * cp + sigm(ai[r] + b_i) * tanhf(ag[r] + b_g);
    float hh = sigm(ao[r] + b_o) * tanhf(cc);
    if (out_bf) out_bf[idx] = (bf16_t)hh;
    else out_f[idx] = hh * mask[mb + r];
  }
}

// ---------------- driver ----------------

extern "C" void kernel_launch(void* const* d_in, const int* in_sizes, int n_in,
                              void* d_out, int out_size, void* d_ws, size_t ws_size,
                              hipStream_t stream) {
  (void)in_sizes; (void)n_in; (void)out_size; (void)ws_size;
  const float* x    = (const float*)d_in[0];
  const int*   ei   = (const int*)d_in[1];
  const float* ew   = (const float*)d_in[2];
  const float* mask = (const float*)d_in[3];
  const float* W1   = (const float*)d_in[4];
  const float* b1   = (const float*)d_in[5];
  const float* W2   = (const float*)d_in[6];
  const float* b2   = (const float*)d_in[7];
  const float* Wih  = (const float*)d_in[8];
  const float* Whh  = (const float*)d_in[9];
  const float* bih  = (const float*)d_in[10];
  const float* bhh  = (const float*)d_in[11];
  const int* srcI = ei;
  const int* dstI = ei + NE;

  unsigned char* p = (unsigned char*)d_ws;
  auto take = [&](size_t bytes) -> void* {
    void* r = (void*)p;
    p += (bytes + 255) & ~(size_t)255;
    return r;
  };
  bf16_t* W1b  = (bf16_t*)take((size_t)HD * HD * 2);
  bf16_t* W2b  = (bf16_t*)take((size_t)HD * HD * 2);
  bf16_t* Wihb = (bf16_t*)take((size_t)4 * HD * HD * 2);
  bf16_t* Whhb = (bf16_t*)take((size_t)4 * HD * HD * 2);
  float*  bb   = (float*)take((size_t)4 * HD * 4);
  float*  deg  = (float*)take((size_t)NN * 4);
  float*  dinv = (float*)take((size_t)NN * 4);
  bf16_t* Xbf  = (bf16_t*)take((size_t)NN * HD * 2);  // layer input bf16 (x1 in-place)
  float*  Hlin = (float*)take((size_t)NN * HD * 4);   // GCN lin output
  float*  Mid  = (float*)take((size_t)NN * HD * 4);   // scatter output (x_mid)
  bf16_t* Mbf  = (bf16_t*)take((size_t)NN * HD * 2);  // x_mid bf16
  float*  C1   = (float*)take((size_t)NN * HD * 4);   // lstm cell state step1
  bf16_t* H1b  = (bf16_t*)take((size_t)NN * HD * 2);  // lstm hidden step1 (bf16)

  const dim3 blk(256);
  const dim3 gGemm(NN / 64, HD / 32);

  // prep: weights -> bf16, bias sum, symmetric-norm dinv
  cvt_f32_to_bf16<<<(HD * HD / 4 + 255) / 256, blk, 0, stream>>>(W1, W1b, HD * HD / 4);
  cvt_f32_to_bf16<<<(HD * HD / 4 + 255) / 256, blk, 0, stream>>>(W2, W2b, HD * HD / 4);
  cvt_f32_to_bf16<<<(4 * HD * HD / 4 + 255) / 256, blk, 0, stream>>>(Wih, Wihb, 4 * HD * HD / 4);
  cvt_f32_to_bf16<<<(4 * HD * HD / 4 + 255) / 256, blk, 0, stream>>>(Whh, Whhb, 4 * HD * HD / 4);
  bias_sum<<<(4 * HD + 255) / 256, blk, 0, stream>>>(bih, bhh, bb);
  deg_init<<<NN / 256, blk, 0, stream>>>(deg);
  deg_edges<<<NE / 256, blk, 0, stream>>>(dstI, ew, deg);
  deg_fin<<<NN / 256, blk, 0, stream>>>(deg, dinv);
  cvt_f32_to_bf16<<<NN * HD / 4 / 256, blk, 0, stream>>>(x, Xbf, NN * HD / 4);

  // ---- layer 1 ----
  gemm_lin<<<gGemm, blk, 0, stream>>>(Xbf, W1b, Hlin);
  scatter_init<<<NN * HD / 256, blk, 0, stream>>>(Hlin, dinv, b1, Mid);
  scatter_edges<<<NE / 4, blk, 0, stream>>>(srcI, dstI, ew, dinv, Hlin, Mid);
  cvt_f32_to_bf16<<<NN * HD / 4 / 256, blk, 0, stream>>>(Mid, Mbf, NN * HD / 4);
  lstm_step1<<<gGemm, blk, 0, stream>>>(Xbf, Wihb, bb, C1, H1b);
  lstm_step2<<<gGemm, blk, 0, stream>>>(Mbf, H1b, Wihb, Whhb, bb, C1,
                                        /*out_bf=*/Xbf, /*out_f=*/nullptr, nullptr);

  // ---- layer 2 (Xbf now holds x1 in bf16) ----
  gemm_lin<<<gGemm, blk, 0, stream>>>(Xbf, W2b, Hlin);
  scatter_init<<<NN * HD / 256, blk, 0, stream>>>(Hlin, dinv, b2, Mid);
  scatter_edges<<<NE / 4, blk, 0, stream>>>(srcI, dstI, ew, dinv, Hlin, Mid);
  cvt_f32_to_bf16<<<NN * HD / 4 / 256, blk, 0, stream>>>(Mid, Mbf, NN * HD / 4);
  lstm_step1<<<gGemm, blk, 0, stream>>>(Xbf, Wihb, bb, C1, H1b);
  lstm_step2<<<gGemm, blk, 0, stream>>>(Mbf, H1b, Wihb, Whhb, bb, C1,
                                        /*out_bf=*/nullptr, /*out_f=*/(float*)d_out, mask);
}
